// QKVAttentionOp_26079041421638
// MI455X (gfx1250) — matrix operand order repaired and checked by reading, not compiled, not run on hardware
//
#include <hip/hip_runtime.h>

typedef __attribute__((ext_vector_type(16))) _Float16 v16h;
typedef __attribute__((ext_vector_type(8)))  _Float16 v8h;
typedef __attribute__((ext_vector_type(8)))  float    v8f;
typedef unsigned int u32;
typedef __attribute__((ext_vector_type(4))) u32 u32x4;
typedef __attribute__((ext_vector_type(8))) int i32x8;
typedef __attribute__((ext_vector_type(4))) int i32x4;

#define WAVES 4
#define BM 64        // q rows per workgroup (16 per wave)
#define BN 64        // keys per iteration
#define HD 64        // head dim
#define LPAD 8       // 8 halfs = 4 dwords of pad per 32-dword row (TDM pad_enable)
#define PPAD 8

__device__ __forceinline__ v8f wmma16(v16h a, v16h b, v8f c) {
  // D = A(16x32 f16) * B(32x16 f16) + C(16x16 f32)
  return __builtin_amdgcn_wmma_f32_16x16x32_f16(false, a, false, b, (short)0, c,
                                                false, false);
}

// build a v16h fragment from two contiguous 8-half (16B aligned) runs
__device__ __forceinline__ v16h ld16(const _Float16* p0, const _Float16* p1) {
  v8h a = *(const v8h*)p0;
  v8h b = *(const v8h*)p1;
  v16h r;
#pragma unroll
  for (int i = 0; i < 8; ++i) { r[i] = a[i]; r[8 + i] = b[i]; }
  return r;
}

// Tensor Data Mover: 2D tile (described in 4-byte units), global -> LDS.
// PAD=true pads LDS dest by 4 dwords every 32 dwords (row stride 144B).
template <bool PAD>
__device__ __forceinline__ void tdm_load_2d(u32 lds_byte, const void* gsrc,
                                            u32 tile0_dw, u32 tile1,
                                            u32 stride_dw, u32 ten0_dw, u32 ten1) {
  const unsigned long long ga = (unsigned long long)gsrc;
  u32x4 g0;
  g0[0] = 1u;                                   // count=1, user descriptor
  g0[1] = lds_byte;                             // lds_addr (bytes)
  g0[2] = (u32)ga;                              // global_addr[31:0]
  g0[3] = (u32)((ga >> 32) & 0x1FFFFFFu) | (2u << 30);  // addr[56:32] | type=2
  i32x8 g1;
  u32 w0 = 2u << 16;                            // data_size = 4B units
  if (PAD) w0 |= (1u << 20) | (4u << 22) | (3u << 25);  // pad 4dw / 32dw
  g1[0] = (int)w0;
  g1[1] = (int)((ten0_dw & 0xFFFFu) << 16);               // tensor_dim0 lo16
  g1[2] = (int)((ten0_dw >> 16) | ((ten1 & 0xFFFFu) << 16)); // dim0 hi | dim1 lo
  g1[3] = (int)((ten1 >> 16) | (tile0_dw << 16));         // dim1 hi | tile_dim0
  g1[4] = (int)(tile1 & 0xFFFFu);                         // tile_dim1 (tile_dim2=0)
  g1[5] = (int)stride_dw;                                 // tensor_dim0_stride lo32
  g1[6] = 0;
  g1[7] = 0;
  i32x4 g2 = {0, 0, 0, 0};
  i32x4 g3 = {0, 0, 0, 0};
  i32x8 g4 = {0, 0, 0, 0, 0, 0, 0, 0};
  __builtin_amdgcn_tensor_load_to_lds(g0, g1, g2, g3, g4, 0);
}

// -------- pre-pass: K -> f16 [BH][S][64], V -> f16 transposed [BH][64][S] ----
__global__ __launch_bounds__(256)
void convert_kv(const float* __restrict__ k, const float* __restrict__ v,
                _Float16* __restrict__ kf, _Float16* __restrict__ vt,
                int S, int H) {
  const int idx = blockIdx.x * 256 + threadIdx.x;  // BH*S*16 threads, 4 elems each
  const int d4 = (idx & 15) * 4;
  const int t = idx >> 4;
  const int key = t % S;
  const int bh = t / S;
  const int b = bh / H, h = bh % H;
  const size_t src = ((size_t)(b * S + key) * H + h) * HD + d4;
  const float4 kq = *(const float4*)(k + src);
  const float4 vq = *(const float4*)(v + src);
  _Float16* kd = kf + ((size_t)bh * S + key) * HD + d4;
  kd[0] = (_Float16)kq.x; kd[1] = (_Float16)kq.y;
  kd[2] = (_Float16)kq.z; kd[3] = (_Float16)kq.w;
  _Float16* vd = vt + (size_t)bh * HD * S + key;
  vd[(size_t)(d4 + 0) * S] = (_Float16)vq.x;
  vd[(size_t)(d4 + 1) * S] = (_Float16)vq.y;
  vd[(size_t)(d4 + 2) * S] = (_Float16)vq.z;
  vd[(size_t)(d4 + 3) * S] = (_Float16)vq.w;
}

// -------- flash attention forward, double-buffered TDM ----------------------
__global__ __launch_bounds__(128)
void fa_fwd_f16wmma(const float* __restrict__ q, const _Float16* __restrict__ kf,
                    const _Float16* __restrict__ vt, const float* __restrict__ kpm,
                    float* __restrict__ out, int S, int Hh) {
  __shared__ _Float16 Ks [2][BN][HD + LPAD];     // K tiles [key][d], TDM-padded
  __shared__ _Float16 Vts[2][HD][BN + LPAD];     // V tiles [d][key], TDM-padded
  __shared__ _Float16 Ps [WAVES][16][BN + PPAD]; // per-wave probabilities
  __shared__ float    Ms [2048];                 // additive mask row (S floats)

  const int tid  = threadIdx.x;
  const int wave = tid >> 5;
  const int lane = tid & 31;
  const int lrow = lane & 15;   // N / M-in-half index
  const int hid  = lane >> 4;   // which 16-lane half

  const int b = blockIdx.z, h = blockIdx.y;
  const int bh = b * Hh + h;
  const int q0 = blockIdx.x * BM + wave * 16;
  const float scale = 0.125f;   // 1/sqrt(64)

  const size_t rowstride = (size_t)Hh * HD;
  const float* qb = q + (size_t)b * S * rowstride + (size_t)h * HD;
  const _Float16* kbase = kf + (size_t)bh * S * HD;       // [key][d]
  const _Float16* vbase = vt + (size_t)bh * HD * S;       // [d][key]

  const u32 ldsK[2] = {(u32)(unsigned long long)(const void*)&Ks[0][0][0],
                       (u32)(unsigned long long)(const void*)&Ks[1][0][0]};
  const u32 ldsV[2] = {(u32)(unsigned long long)(const void*)&Vts[0][0][0],
                       (u32)(unsigned long long)(const void*)&Vts[1][0][0]};
  const u32 ldsM   =  (u32)(unsigned long long)(const void*)&Ms[0];

  // ---- prologue: mask row + first K/V tiles via TDM (wave 0 drives the DMA)
  if (wave == 0) {
    tdm_load_2d<false>(ldsM, kpm + (size_t)b * S, (u32)S, 1u, (u32)S, (u32)S, 1u);
    tdm_load_2d<true >(ldsK[0], kbase, 32u, 64u, 32u, 32u, 64u);
    tdm_load_2d<true >(ldsV[0], vbase, 32u, 64u, (u32)(S / 2), (u32)(S / 2), 64u);
  }

  // Q A-fragments (ISA 16-bit A 16x32: lo lanes K 0-7/16-23, hi lanes 8-15/24-31)
  // — global loads overlap the prologue DMA
  v16h aQ[2];
  {
    const float* qr = qb + (size_t)(q0 + lrow) * rowstride;
#pragma unroll
    for (int c = 0; c < 2; ++c) {
#pragma unroll
      for (int e = 0; e < 8; ++e) {
        aQ[c][e]     = (_Float16)qr[c * 32 + hid * 8 + e];
        aQ[c][8 + e] = (_Float16)qr[c * 32 + 16 + hid * 8 + e];
      }
    }
  }

  v8f O[4] = {};                 // d tile nt: col = nt*16 + lrow
  float mrow[8], lsum[8];
#pragma unroll
  for (int r = 0; r < 8; ++r) { mrow[r] = -__builtin_inff(); lsum[r] = 0.f; }

  if (wave == 0) __builtin_amdgcn_s_wait_tensorcnt(0);
  __syncthreads();

  const int T = S / BN;
  for (int it = 0; it < T; ++it) {
    const int cur = it & 1;
    const int key0 = it * BN;

    // prefetch next tile pair into the other buffer; overlaps with compute below
    if (wave == 0 && it + 1 < T) {
      tdm_load_2d<true>(ldsK[cur ^ 1], kbase + (size_t)(key0 + BN) * HD,
                        32u, 64u, 32u, 32u, 64u);
      tdm_load_2d<true>(ldsV[cur ^ 1], vbase + key0 + BN,
                        32u, 64u, (u32)(S / 2), (u32)(S / 2), 64u);
    }

    // ---- GEMM1: S = Q * K^T, four 16x16 C tiles (keys t*16 .. t*16+15)
    v8f Sc[4] = {};
#pragma unroll
    for (int t = 0; t < 4; ++t) {
#pragma unroll
      for (int c = 0; c < 2; ++c) {
        const _Float16* kp = &Ks[cur][t * 16 + lrow][c * 32 + hid * 8];
        Sc[t] = wmma16(aQ[c], ld16(kp, kp + 16), Sc[t]);
      }
    }

    float mask[4];
#pragma unroll
    for (int t = 0; t < 4; ++t) mask[t] = Ms[key0 + t * 16 + lrow];

    // ---- online softmax (row m = r + 8*hid lives in one 16-lane half)
#pragma unroll
    for (int r = 0; r < 8; ++r) {
      float s[4], mx;
#pragma unroll
      for (int t = 0; t < 4; ++t) s[t] = Sc[t][r] * scale + mask[t];
      mx = fmaxf(fmaxf(s[0], s[1]), fmaxf(s[2], s[3]));
#pragma unroll
      for (int off = 8; off >= 1; off >>= 1)
        mx = fmaxf(mx, __shfl_xor(mx, off, 32));   // stays within 16-lane half
      const float mnew = fmaxf(mrow[r], mx);
      float p[4], ps = 0.f;
#pragma unroll
      for (int t = 0; t < 4; ++t) { p[t] = __expf(s[t] - mnew); ps += p[t]; }
#pragma unroll
      for (int off = 8; off >= 1; off >>= 1)
        ps += __shfl_xor(ps, off, 32);
      const float corr = __expf(mrow[r] - mnew);
      mrow[r] = mnew;
      lsum[r] = lsum[r] * corr + ps;
#pragma unroll
      for (int nt = 0; nt < 4; ++nt) O[nt][r] *= corr;
#pragma unroll
      for (int t = 0; t < 4; ++t)
        Ps[wave][r + 8 * hid][t * 16 + lrow] = (_Float16)p[t];
    }
    // DS ops are in-order per wave; drain before re-reading Ps
    asm volatile("s_wait_dscnt 0" ::: "memory");

    // ---- GEMM2: O += P(16x64) * V(64x64)
    v16h aP[2];
#pragma unroll
    for (int c = 0; c < 2; ++c)
      aP[c] = ld16(&Ps[wave][lrow][c * 32 + hid * 8],
                   &Ps[wave][lrow][c * 32 + 16 + hid * 8]);
#pragma unroll
    for (int nt = 0; nt < 4; ++nt) {
#pragma unroll
      for (int c = 0; c < 2; ++c) {
        const _Float16* vp = &Vts[cur][nt * 16 + lrow][c * 32 + hid * 8];
        O[nt] = wmma16(aP[c], ld16(vp, vp + 16), O[nt]);
      }
    }

    // next tile pair landed; one barrier also fences buffer reuse two steps out
    if (wave == 0) __builtin_amdgcn_s_wait_tensorcnt(0);
    __syncthreads();
  }

  // ---- epilogue: divide by softmax sum, write fp32 [B,S,H,D]
#pragma unroll
  for (int r = 0; r < 8; ++r) {
    const float inv = 1.0f / lsum[r];
    const size_t orow = ((size_t)b * S + (q0 + r + 8 * hid)) * rowstride +
                        (size_t)h * HD;
#pragma unroll
    for (int nt = 0; nt < 4; ++nt)
      out[orow + nt * 16 + lrow] = O[nt][r] * inv;
  }
}

extern "C" void kernel_launch(void* const* d_in, const int* in_sizes, int n_in,
                              void* d_out, int out_size, void* d_ws, size_t ws_size,
                              hipStream_t stream) {
  const float* q   = (const float*)d_in[0];
  const float* k   = (const float*)d_in[1];
  const float* v   = (const float*)d_in[2];
  const float* kpm = (const float*)d_in[3];
  float* out = (float*)d_out;

  const int B = 2, S = 2048;
  const int H = in_sizes[0] / (in_sizes[3] * HD);   // q = B*S*H*64, mask = B*S
  const int BH = B * H;

  _Float16* kf = (_Float16*)d_ws;                   // [BH][S][64]
  _Float16* vt = kf + (size_t)BH * S * HD;          // [BH][64][S]

  const int cvt_threads = BH * S * 16;              // 4 elems per thread
  convert_kv<<<cvt_threads / 256, 256, 0, stream>>>(k, v, kf, vt, S, H);

  dim3 grid(S / BM, H, B);
  fa_fwd_f16wmma<<<grid, 128, 0, stream>>>(q, kf, vt, kpm, out, S, H);
}
